// GNNModel_73160472920253
// MI455X (gfx1250) — compile-verified, matrix operand order checked by reading, not run on hardware
//
#include <hip/hip_runtime.h>

typedef _Float16 half_t;
typedef __attribute__((ext_vector_type(16))) _Float16 v16h;
typedef __attribute__((ext_vector_type(8)))  _Float16 v8h;
typedef __attribute__((ext_vector_type(8)))  float    v8f;

__device__ __forceinline__ void atomAddF(float* p, float v) {
  __hip_atomic_fetch_add(p, v, __ATOMIC_RELAXED, __HIP_MEMORY_SCOPE_AGENT);
}

// ---------------------------------------------------------------------------
// Pre-swizzle B[K x 64] (f32) into per-lane WMMA operand layout (f16):
//   out[((kcIdx*4 + nt)*32 + lane)*16 + j]
//   lane l: n = nt*16 + (l&15), koff = (l>>4)*8
//   j 0..7  -> K = kcIdx*32 + koff + j
//   j 8..15 -> K = kcIdx*32 + koff + 16 + (j-8)
// so the GEMM loads one contiguous 32B v16h per lane per tile.
// ---------------------------------------------------------------------------
__global__ void swizzle_b_kernel(const float* __restrict__ W, half_t* __restrict__ out, int K) {
  int o = blockIdx.x * blockDim.x + threadIdx.x;
  int total = (K >> 5) * 4 * 32 * 16;
  if (o >= total) return;
  int j     = o & 15;
  int lane  = (o >> 4) & 31;
  int nt    = (o >> 9) & 3;
  int kcIdx = o >> 11;
  int n    = nt * 16 + (lane & 15);
  int koff = (lane >> 4) * 8;
  int k    = kcIdx * 32 + koff + ((j < 8) ? j : (8 + j));  // j>=8 -> +16+(j-8)
  out[o] = (half_t)W[(size_t)k * 64 + n];
}

// ---------------------------------------------------------------------------
// Degree over dst (self loop added later as +1)
// ---------------------------------------------------------------------------
__global__ void degree_kernel(const int* __restrict__ ei, float* __restrict__ deg, int E) {
  int e = blockIdx.x * blockDim.x + threadIdx.x;
  if (e >= E) return;
  atomAddF(deg + ei[E + e], 1.0f);
}

__global__ void dinv_kernel(float* __restrict__ dinv, int N) {
  int n = blockIdx.x * blockDim.x + threadIdx.x;
  if (n >= N) return;
  dinv[n] = rsqrtf(dinv[n] + 1.0f);  // deg >= 1 incl. self-loop
}

// ---------------------------------------------------------------------------
// xw1 = x @ W1  (N x 6) @ (6 x 64): one thread per (node, channel)
// ---------------------------------------------------------------------------
__global__ void xw1_kernel(const float* __restrict__ x, const float* __restrict__ W1,
                           float* __restrict__ out, int N) {
  int tid = blockIdx.x * blockDim.x + threadIdx.x;
  if (tid >= N * 64) return;
  int n = tid >> 6, c = tid & 63;
  const float* xr = x + (size_t)n * 6;
  float s = 0.0f;
#pragma unroll
  for (int k = 0; k < 6; ++k) s += xr[k] * W1[k * 64 + c];
  out[tid] = s;
}

// ---------------------------------------------------------------------------
// Edge scatter: out[dst] += xw[src] * dinv[src]*dinv[dst]
// 32 lanes per edge, 2 channels per lane (64 channels)
// ---------------------------------------------------------------------------
__global__ void edge_aggregate_kernel(const int* __restrict__ ei, const float* __restrict__ xw,
                                      const float* __restrict__ dinv, float* __restrict__ out, int E) {
  int gtid = blockIdx.x * blockDim.x + threadIdx.x;
  int e = gtid >> 5;
  if (e >= E) return;
  int lane = gtid & 31;
  int s = ei[e];
  int d = ei[E + e];
  float norm = dinv[s] * dinv[d];
  int c = lane << 1;
  const float2 v = *(const float2*)(xw + (size_t)s * 64 + c);
  float* o = out + (size_t)d * 64 + c;
  atomAddF(o, v.x * norm);
  atomAddF(o + 1, v.y * norm);
}

// ---------------------------------------------------------------------------
// Layer-1 finalize: h = relu(agg + xw*dinv^2 + b)  -> f16 for WMMA layer 2
// ---------------------------------------------------------------------------
__global__ void finalize1_kernel(const float* __restrict__ agg, const float* __restrict__ xw,
                                 const float* __restrict__ dinv, const float* __restrict__ b,
                                 half_t* __restrict__ outh, int N) {
  int tid = blockIdx.x * blockDim.x + threadIdx.x;
  if (tid >= N * 64) return;
  int n = tid >> 6, c = tid & 63;
  float di = dinv[n];
  float v = agg[tid] + xw[tid] * di * di + b[c];
  outh[tid] = (half_t)fmaxf(v, 0.0f);
}

// ---------------------------------------------------------------------------
// WMMA GEMM: D[M x 64] = A[M x K](f16) @ Bsw(pre-swizzled f16), bias+relu opt.
// One wave per 16-row tile; M must be a multiple of 16 (both call sites are).
// A layout per CDNA5 ISA 7.12.2 (two 16B loads/lane); B is one 32B load/lane.
// C/D f32 16x16: VGPR r -> row r (lanes 0-15) / r+8 (lanes 16-31), N = l&15.
// ---------------------------------------------------------------------------
__global__ __launch_bounds__(256)
void wmma_gemm_f16n64_kernel(const half_t* __restrict__ A, const half_t* __restrict__ Bsw,
                             const float* __restrict__ bias, float* __restrict__ D,
                             int M, int K, int relu) {
  int wave = (blockIdx.x * blockDim.x + threadIdx.x) >> 5;
  int lane = threadIdx.x & 31;
  int rowbase = wave << 4;
  if (rowbase >= M) return;             // wave-uniform: EXEC stays all-ones
  int halfsel = lane >> 4;
  int l15 = lane & 15;
  int mload = rowbase + l15;
  int koff = halfsel * 8;

  v8f acc[4] = {};
  for (int kc = 0; kc < K; kc += 32) {
    const v8h a0 = *(const v8h*)(A + (size_t)mload * K + kc + koff);
    const v8h a1 = *(const v8h*)(A + (size_t)mload * K + kc + koff + 16);
    v16h a;
#pragma unroll
    for (int j = 0; j < 8; ++j) { a[j] = a0[j]; a[8 + j] = a1[j]; }
    int kcIdx = kc >> 5;
#pragma unroll
    for (int nt = 0; nt < 4; ++nt) {
      const v16h b = *(const v16h*)(Bsw + ((size_t)(kcIdx * 4 + nt) * 32 + lane) * 16);
      acc[nt] = __builtin_amdgcn_wmma_f32_16x16x32_f16(
          /*neg_a=*/false, a, /*neg_b=*/false, b,
          /*c_mod=*/(short)0, acc[nt], /*reuse_a=*/false, /*reuse_b=*/false);
    }
  }

  int mstore = rowbase + halfsel * 8;
#pragma unroll
  for (int nt = 0; nt < 4; ++nt) {
    int n = nt * 16 + l15;
    float bv = bias ? bias[n] : 0.0f;
#pragma unroll
    for (int r = 0; r < 8; ++r) {
      float v = acc[nt][r] + bv;
      if (relu) v = fmaxf(v, 0.0f);
      D[(size_t)(mstore + r) * 64 + n] = v;
    }
  }
}

// ---------------------------------------------------------------------------
// Layer-2 finalize + mean-pool accumulation
// ---------------------------------------------------------------------------
__global__ void finalize2_pool_kernel(const float* __restrict__ agg, const float* __restrict__ xw,
                                      const float* __restrict__ dinv, const float* __restrict__ b,
                                      const int* __restrict__ batch,
                                      float* __restrict__ gsum, float* __restrict__ gcnt, int N) {
  int tid = blockIdx.x * blockDim.x + threadIdx.x;
  if (tid >= N * 64) return;
  int n = tid >> 6, c = tid & 63;
  float di = dinv[n];
  float v = fmaxf(agg[tid] + xw[tid] * di * di + b[c], 0.0f);
  int g = batch[n];
  atomAddF(gsum + (size_t)g * 64 + c, v);
  if (c == 0) atomAddF(gcnt + g, 1.0f);
}

// ---------------------------------------------------------------------------
// Build concatenated MLP input [pooled(64) | lig(16) | add(16) | base(16) | aryl(16)] -> f16
// ---------------------------------------------------------------------------
__global__ void cat_kernel(const float* __restrict__ gsum, const float* __restrict__ gcnt,
                           const int* __restrict__ lig, const int* __restrict__ add,
                           const int* __restrict__ bas, const int* __restrict__ ary,
                           const float* __restrict__ eL, const float* __restrict__ eA,
                           const float* __restrict__ eB, const float* __restrict__ eY,
                           half_t* __restrict__ cat, int G) {
  int tid = blockIdx.x * blockDim.x + threadIdx.x;
  if (tid >= G * 128) return;
  int g = tid >> 7, c = tid & 127;
  float v;
  if (c < 64)       v = gsum[(size_t)g * 64 + c] / fmaxf(gcnt[g], 1.0f);
  else if (c < 80)  v = eL[lig[g] * 16 + (c - 64)];
  else if (c < 96)  v = eA[add[g] * 16 + (c - 80)];
  else if (c < 112) v = eB[bas[g] * 16 + (c - 96)];
  else              v = eY[ary[g] * 16 + (c - 112)];
  cat[tid] = (half_t)v;
}

// ---------------------------------------------------------------------------
// out[g] = hmid[g,:] . lin2_W + lin2_b
// ---------------------------------------------------------------------------
__global__ void final_kernel(const float* __restrict__ hmid, const float* __restrict__ w,
                             const float* __restrict__ b, float* __restrict__ out, int G) {
  int g = blockIdx.x * blockDim.x + threadIdx.x;
  if (g >= G) return;
  float s = b[0];
#pragma unroll
  for (int c = 0; c < 64; ++c) s += hmid[(size_t)g * 64 + c] * w[c];
  out[g] = s;
}

// ---------------------------------------------------------------------------
extern "C" void kernel_launch(void* const* d_in, const int* in_sizes, int n_in,
                              void* d_out, int out_size, void* d_ws, size_t ws_size,
                              hipStream_t stream) {
  const float* x     = (const float*)d_in[0];
  const int*   ei    = (const int*)d_in[1];
  const int*   batch = (const int*)d_in[2];
  const int*   ligI  = (const int*)d_in[3];
  const int*   addI  = (const int*)d_in[4];
  const int*   basI  = (const int*)d_in[5];
  const int*   aryI  = (const int*)d_in[6];
  const float* embL  = (const float*)d_in[7];
  const float* embA  = (const float*)d_in[8];
  const float* embB  = (const float*)d_in[9];
  const float* embY  = (const float*)d_in[10];
  const float* W1    = (const float*)d_in[11];
  const float* b1    = (const float*)d_in[12];
  const float* W2    = (const float*)d_in[13];
  const float* b2    = (const float*)d_in[14];
  const float* l1W   = (const float*)d_in[15];
  const float* l1b   = (const float*)d_in[16];
  const float* l2W   = (const float*)d_in[17];
  const float* l2b   = (const float*)d_in[18];
  float* out = (float*)d_out;
  (void)n_in; (void)out_size; (void)ws_size;

  const int N = in_sizes[0] / 6;
  const int E = in_sizes[1] / 2;
  const int G = in_sizes[3];
  const size_t NC = (size_t)N * 64;

  // Workspace carve (256-B aligned). bufA: xw1 then xw2; bufB: agg1 then agg2.
  char* wsb = (char*)d_ws;
  size_t off = 0;
  auto carve = [&](size_t bytes) -> char* {
    char* p = wsb + off;
    off = (off + bytes + 255) & ~(size_t)255;
    return p;
  };
  float*  bufA  = (float*)carve(NC * 4);
  float*  bufB  = (float*)carve(NC * 4);
  half_t* h1h   = (half_t*)carve(NC * 2);
  float*  dinv  = (float*)carve((size_t)N * 4);
  half_t* W2sw  = (half_t*)carve(2 * 4 * 32 * 16 * 2);   // K=64 swizzled
  half_t* l1Wsw = (half_t*)carve(4 * 4 * 32 * 16 * 2);   // K=128 swizzled
  float*  gsum  = (float*)carve((size_t)G * 64 * 4);
  float*  gcnt  = (float*)carve((size_t)G * 4);
  half_t* cath  = (half_t*)carve((size_t)G * 128 * 2);
  float*  hmid  = (float*)carve((size_t)G * 64 * 4);

  const int T = 256;
  const int blksNC   = (int)((NC + T - 1) / T);
  const int blksEdge = (int)(((size_t)E * 32 + T - 1) / T);

  // --- weights -> swizzled f16, degrees ---
  hipMemsetAsync(dinv, 0, (size_t)N * 4, stream);
  swizzle_b_kernel<<<(4096 + T - 1) / T, T, 0, stream>>>(W2, W2sw, 64);
  swizzle_b_kernel<<<(8192 + T - 1) / T, T, 0, stream>>>(l1W, l1Wsw, 128);
  degree_kernel<<<(E + T - 1) / T, T, 0, stream>>>(ei, dinv, E);
  dinv_kernel<<<(N + T - 1) / T, T, 0, stream>>>(dinv, N);

  // --- GCN layer 1 ---
  xw1_kernel<<<blksNC, T, 0, stream>>>(x, W1, bufA, N);
  hipMemsetAsync(bufB, 0, NC * 4, stream);
  edge_aggregate_kernel<<<blksEdge, T, 0, stream>>>(ei, bufA, dinv, bufB, E);
  finalize1_kernel<<<blksNC, T, 0, stream>>>(bufB, bufA, dinv, b1, h1h, N);

  // --- GCN layer 2: xw2 = h1 @ W2 via WMMA ---
  {
    int waves = (N + 15) / 16;
    int blks = (int)(((size_t)waves * 32 + T - 1) / T);
    wmma_gemm_f16n64_kernel<<<blks, T, 0, stream>>>(h1h, W2sw, nullptr, bufA, N, 64, 0);
  }
  hipMemsetAsync(bufB, 0, NC * 4, stream);
  edge_aggregate_kernel<<<blksEdge, T, 0, stream>>>(ei, bufA, dinv, bufB, E);

  // --- finalize + mean pool ---
  hipMemsetAsync(gsum, 0, (size_t)G * 64 * 4, stream);
  hipMemsetAsync(gcnt, 0, (size_t)G * 4, stream);
  finalize2_pool_kernel<<<blksNC, T, 0, stream>>>(bufB, bufA, dinv, b2, batch, gsum, gcnt, N);

  // --- MLP ---
  cat_kernel<<<(G * 128 + T - 1) / T, T, 0, stream>>>(gsum, gcnt, ligI, addI, basI, aryI,
                                                      embL, embA, embB, embY, cath, G);
  {
    int waves = (G + 15) / 16;
    int blks = (int)(((size_t)waves * 32 + T - 1) / T);
    wmma_gemm_f16n64_kernel<<<blks, T, 0, stream>>>(cath, l1Wsw, l1b, hmid, G, 128, 1);
  }
  final_kernel<<<(G + T - 1) / T, T, 0, stream>>>(hmid, l2W, l2b, out, G);
}
